// MutiHeadAttention_26353919328588
// MI455X (gfx1250) — matrix operand order repaired and checked by reading, not compiled, not run on hardware
//
#include <hip/hip_runtime.h>

// ---------------------------------------------------------------------------
// MHA forward for MI455X (gfx1250), wave32, bf16 WMMA with f32 accumulate.
// ---------------------------------------------------------------------------

typedef __attribute__((ext_vector_type(16))) __bf16 bf16x16;
typedef __attribute__((ext_vector_type(8)))  __bf16 bf16x8;
typedef __attribute__((ext_vector_type(8)))  float  f32x8;

union BFrag {
    bf16x16        v;
    bf16x8         h[2];
    unsigned short u[16];
};

__device__ __forceinline__ unsigned short f32_to_bf16(float f) {
    unsigned int u = __float_as_uint(f);
    u += 0x7FFFu + ((u >> 16) & 1u);   // round-to-nearest-even
    return (unsigned short)(u >> 16);
}

static constexpr int BATCH = 8;
static constexpr int S     = 1024;
static constexpr int H     = 1024;
static constexpr int NH    = 16;
static constexpr int HD    = 64;
static constexpr int MTOK  = BATCH * S;   // 8192 rows

// ---------------------------------------------------------------------------
// 1) fp32 -> bf16 convert
// ---------------------------------------------------------------------------
__global__ __launch_bounds__(256)
void mha_cvt_bf16_kernel(const float* __restrict__ src,
                         unsigned short* __restrict__ dst, int n) {
    int i      = blockIdx.x * blockDim.x + threadIdx.x;
    int stride = gridDim.x * blockDim.x;
    for (; i < n; i += stride) dst[i] = f32_to_bf16(src[i]);
}

// ---------------------------------------------------------------------------
// 2) QKV projection: y = x@W + b, then RoPE (q,k) + 1/sqrt(HD) on q.
//    q,k scatter to head-major bf16 [B, NH, S, HD].
//    v scatter TRANSPOSED to d-major bf16 [B, NH, HD, S] so the PV GEMM's
//    B-fragments (K = key) are contiguous 16B loads.
//    Block: 256 threads (8 waves). Tile: 128 rows x 64 cols (one head).
//    grid = (H/64, MTOK/128, 3)   z: 0=q 1=k 2=v
// ---------------------------------------------------------------------------
__global__ __launch_bounds__(256)
void mha_qkv_gemm_rope_kernel(const unsigned short* __restrict__ xb,
                              const unsigned short* __restrict__ Wqb,
                              const unsigned short* __restrict__ Wkb,
                              const unsigned short* __restrict__ Wvb,
                              const float* __restrict__ bq,
                              const float* __restrict__ bk,
                              const float* __restrict__ bv,
                              const int*   __restrict__ pos_idx,
                              unsigned short* __restrict__ qh,
                              unsigned short* __restrict__ kh,
                              unsigned short* __restrict__ vht) {
    __shared__ unsigned short As[128][32];   // 8 KB   (M x K tile)
    __shared__ unsigned short Bst[64][32];   // 4 KB   (N x K, transposed)

    const int z = blockIdx.z;
    const unsigned short* W    = (z == 0) ? Wqb : (z == 1) ? Wkb : Wvb;
    const float*          bias = (z == 0) ? bq  : (z == 1) ? bk  : bv;
    unsigned short*       dst  = (z == 0) ? qh  : (z == 1) ? kh  : vht;

    const int n0  = blockIdx.x * 64;
    const int m0  = blockIdx.y * 128;
    const int tid = threadIdx.x;
    const int wid = tid >> 5, lane = tid & 31;
    const int l16 = lane & 15, hi = lane >> 4;

    f32x8 acc[4];
    for (int i = 0; i < 4; ++i)
        for (int e = 0; e < 8; ++e) acc[i][e] = 0.0f;

    for (int kk = 0; kk < H; kk += 32) {
        // A tile: 128x32, 2 x 16B vector copies per thread
        {
            const int r = tid >> 1, half = tid & 1;
            const uint4* sp = (const uint4*)(xb + (size_t)(m0 + r) * H + kk + half * 16);
            uint4* dp = (uint4*)&As[r][half * 16];
            dp[0] = sp[0];
            dp[1] = sp[1];
        }
        // B tile: read W row-major coalesced, store transposed (N-major)
        {
            const int kr = tid >> 3, c = tid & 7;
            const unsigned short* sp = W + (size_t)(kk + kr) * H + n0 + c * 8;
#pragma unroll
            for (int j = 0; j < 8; ++j) Bst[c * 8 + j][kr] = sp[j];
        }
        __syncthreads();

        BFrag a;   // A 16x32: lanes 0-15 K-groups {0..7,16..23}, lanes 16-31 {8..15,24..31}
        a.h[0] = *(const bf16x8*)&As[wid * 16 + l16][hi * 8];
        a.h[1] = *(const bf16x8*)&As[wid * 16 + l16][hi * 8 + 16];
#pragma unroll
        for (int ns = 0; ns < 4; ++ns) {
            BFrag b;   // B 32x16: lanes 0-15 K-groups {0..7,16..23}, lanes 16-31 {8..15,24..31}
            b.h[0] = *(const bf16x8*)&Bst[ns * 16 + l16][hi * 8];
            b.h[1] = *(const bf16x8*)&Bst[ns * 16 + l16][hi * 8 + 16];
            acc[ns] = __builtin_amdgcn_wmma_f32_16x16x32_bf16(
                false, a.v, false, b.v, (short)0, acc[ns], false, false);
        }
        __syncthreads();
    }

    // bias (pre-RoPE, as in reference)
#pragma unroll
    for (int ns = 0; ns < 4; ++ns) {
        const float bvv = bias[n0 + ns * 16 + l16];
#pragma unroll
        for (int r = 0; r < 8; ++r) acc[ns][r] += bvv;
    }

    const int head = n0 >> 6;   // N tile == one head

    if (z < 2) {
        // RoPE, faithful to the precedence bug: inv_freq[i] = HD / 10000^(2i)
        // (overflows to inf -> theta = 0 for large i, same as jnp float32)
        const float j0 = 2.0f * (float)l16;         // ns even: i = l16
        const float j1 = 2.0f * (float)(16 + l16);  // ns odd : i = 16 + l16
        const float invf0 = 64.0f / powf(10000.0f, j0);
        const float invf1 = 64.0f / powf(10000.0f, j1);
        const float qscale = (z == 0) ? 0.125f : 1.0f;   // fold 1/sqrt(HD) into q
#pragma unroll
        for (int r = 0; r < 8; ++r) {
            const int m = m0 + wid * 16 + r + hi * 8;
            const int s = m & (S - 1);
            const float pos = (float)pos_idx[s];
            const float t0 = pos * invf0, t1 = pos * invf1;
            const float c0 = cosf(t0), s0 = sinf(t0);
            const float c1 = cosf(t1), s1 = sinf(t1);
            const float x0 = acc[0][r], x1 = acc[1][r];
            const float x2 = acc[2][r], x3 = acc[3][r];
            acc[0][r] = (x0 * c0 - x2 * s0) * qscale;   // d < 32: -x[d+32]*sin
            acc[1][r] = (x1 * c1 - x3 * s1) * qscale;
            acc[2][r] = (x2 * c0 + x0 * s0) * qscale;   // d >= 32: +x[d-32]*sin
            acc[3][r] = (x3 * c1 + x1 * s1) * qscale;
        }
    }

    // scatter
#pragma unroll
    for (int r = 0; r < 8; ++r) {
        const int m  = m0 + wid * 16 + r + hi * 8;
        const int bb = m >> 10;
        const int s  = m & (S - 1);
        const size_t hb = ((size_t)(bb * NH + head)) * (size_t)(S * HD);
        if (z == 2) {
            // v transposed: [B, NH, HD, S]
#pragma unroll
            for (int ns = 0; ns < 4; ++ns)
                dst[hb + (size_t)(ns * 16 + l16) * S + s] = f32_to_bf16(acc[ns][r]);
        } else {
            // q/k head-major: [B, NH, S, HD]
#pragma unroll
            for (int ns = 0; ns < 4; ++ns)
                dst[hb + (size_t)s * HD + ns * 16 + l16] = f32_to_bf16(acc[ns][r]);
        }
    }
}

// ---------------------------------------------------------------------------
// 3) Attention: per (b, h, 16-row q tile). Block: 128 threads (4 waves).
//    Phase 1: scores = q @ k^T into 64KB LDS (waves split the 1024 keys).
//    Phase 2: full-row softmax (float4 passes, shuffle reductions, in place).
//    Phase 3: out = P @ V^T (waves split the 64 output dims; V is d-major so
//             the B-fragment is two contiguous 16B loads).
//    grid = (S/16, NH, B)
// ---------------------------------------------------------------------------
__global__ __launch_bounds__(128)
void mha_attn_kernel(const unsigned short* __restrict__ qh,
                     const unsigned short* __restrict__ kh,
                     const unsigned short* __restrict__ vht,
                     unsigned short* __restrict__ attnb) {
    __shared__ float sc[16][1024];   // exactly 64 KB

    const int qt = blockIdx.x, head = blockIdx.y, bb = blockIdx.z;
    const size_t hoff = ((size_t)(bb * NH + head)) * S * HD;
    const unsigned short* qp = qh + hoff;
    const unsigned short* kp = kh + hoff;
    const unsigned short* vp = vht + hoff;   // [HD][S] within the head

    const int tid = threadIdx.x, wid = tid >> 5, lane = tid & 31;
    const int l16 = lane & 15, hi = lane >> 4;

    // ---- Phase 1: scores (q pre-scaled by 1/sqrt(HD)) ----
    BFrag aq[2];   // q rows contiguous (HD=64): pure 16B vector loads
    {
        const unsigned short* base = qp + (size_t)(qt * 16 + l16) * HD + hi * 8;
        aq[0].h[0] = *(const bf16x8*)(base);
        aq[0].h[1] = *(const bf16x8*)(base + 16);
        aq[1].h[0] = *(const bf16x8*)(base + 32);
        aq[1].h[1] = *(const bf16x8*)(base + 48);
    }
    for (int kt = 0; kt < 16; ++kt) {
        const int key0 = wid * 256 + kt * 16;
        f32x8 acc;
        for (int e = 0; e < 8; ++e) acc[e] = 0.0f;
#pragma unroll
        for (int ds = 0; ds < 2; ++ds) {
            BFrag bk;  // k^T fragment: column = key, K = d contiguous in memory
            const unsigned short* base = kp + (size_t)(key0 + l16) * HD + ds * 32 + hi * 8;
            bk.h[0] = *(const bf16x8*)(base);
            bk.h[1] = *(const bf16x8*)(base + 16);
            acc = __builtin_amdgcn_wmma_f32_16x16x32_bf16(
                false, aq[ds].v, false, bk.v, (short)0, acc, false, false);
        }
#pragma unroll
        for (int r = 0; r < 8; ++r) sc[r + hi * 8][key0 + l16] = acc[r];
    }
    __syncthreads();

    // ---- Phase 2: softmax over each full row; thread-local 128-col segment ----
    {
        const int row = tid >> 3, seg = tid & 7;   // 8 consecutive lanes per row
        float4* rp4 = (float4*)&sc[row][seg * 128];
        float mx = -3.0e38f;
        for (int c = 0; c < 32; ++c) {
            const float4 t = rp4[c];
            mx = fmaxf(mx, fmaxf(fmaxf(t.x, t.y), fmaxf(t.z, t.w)));
        }
        for (int msk = 1; msk < 8; msk <<= 1) mx = fmaxf(mx, __shfl_xor(mx, msk, 32));
        float sum = 0.0f;
        for (int c = 0; c < 32; ++c) {
            float4 t = rp4[c];
            t.x = __expf(t.x - mx);
            t.y = __expf(t.y - mx);
            t.z = __expf(t.z - mx);
            t.w = __expf(t.w - mx);
            rp4[c] = t;
            sum += (t.x + t.y) + (t.z + t.w);
        }
        for (int msk = 1; msk < 8; msk <<= 1) sum += __shfl_xor(sum, msk, 32);
        const float inv = 1.0f / sum;
        for (int c = 0; c < 32; ++c) {
            float4 t = rp4[c];
            t.x *= inv; t.y *= inv; t.z *= inv; t.w *= inv;
            rp4[c] = t;
        }
    }
    __syncthreads();

    // ---- Phase 3: out = P @ V^T, wave wid owns output dims [wid*16, wid*16+16) ----
    f32x8 oacc;
    for (int e = 0; e < 8; ++e) oacc[e] = 0.0f;
    for (int ks = 0; ks < 32; ++ks) {
        BFrag pa;   // P fragment from LDS f32, converted to bf16 on the fly
        const float* p0 = &sc[l16][ks * 32 + hi * 8];
#pragma unroll
        for (int e = 0; e < 8; ++e) pa.u[e] = f32_to_bf16(p0[e]);
#pragma unroll
        for (int e = 0; e < 8; ++e) pa.u[8 + e] = f32_to_bf16(p0[e + 16]);
        BFrag vb;   // V fragment: column = d (row of v^T), K = key contiguous
        const unsigned short* vbase = vp + (size_t)(wid * 16 + l16) * S + ks * 32 + hi * 8;
        vb.h[0] = *(const bf16x8*)(vbase);
        vb.h[1] = *(const bf16x8*)(vbase + 16);
        oacc = __builtin_amdgcn_wmma_f32_16x16x32_bf16(
            false, pa.v, false, vb.v, (short)0, oacc, false, false);
    }
#pragma unroll
    for (int r = 0; r < 8; ++r) {
        const int qr = qt * 16 + r + hi * 8;
        const size_t off = ((size_t)(bb * S + qr)) * H + head * HD + wid * 16 + l16;
        attnb[off] = f32_to_bf16(oacc[r]);
    }
}

// ---------------------------------------------------------------------------
// 4) Output projection: out = attn @ Wo + bo, fp32 output.
//    grid = (H/64, MTOK/128)
// ---------------------------------------------------------------------------
__global__ __launch_bounds__(256)
void mha_out_gemm_kernel(const unsigned short* __restrict__ ab,
                         const unsigned short* __restrict__ Wob,
                         const float* __restrict__ bo,
                         float* __restrict__ out) {
    __shared__ unsigned short As[128][32];
    __shared__ unsigned short Bst[64][32];

    const int n0  = blockIdx.x * 64;
    const int m0  = blockIdx.y * 128;
    const int tid = threadIdx.x;
    const int wid = tid >> 5, lane = tid & 31;
    const int l16 = lane & 15, hi = lane >> 4;

    f32x8 acc[4];
    for (int i = 0; i < 4; ++i)
        for (int e = 0; e < 8; ++e) acc[i][e] = 0.0f;

    for (int kk = 0; kk < H; kk += 32) {
        {
            const int r = tid >> 1, half = tid & 1;
            const uint4* sp = (const uint4*)(ab + (size_t)(m0 + r) * H + kk + half * 16);
            uint4* dp = (uint4*)&As[r][half * 16];
            dp[0] = sp[0];
            dp[1] = sp[1];
        }
        {
            const int kr = tid >> 3, c = tid & 7;
            const unsigned short* sp = Wob + (size_t)(kk + kr) * H + n0 + c * 8;
#pragma unroll
            for (int j = 0; j < 8; ++j) Bst[c * 8 + j][kr] = sp[j];
        }
        __syncthreads();

        BFrag a;
        a.h[0] = *(const bf16x8*)&As[wid * 16 + l16][hi * 8];
        a.h[1] = *(const bf16x8*)&As[wid * 16 + l16][hi * 8 + 16];
#pragma unroll
        for (int ns = 0; ns < 4; ++ns) {
            BFrag b;
            b.h[0] = *(const bf16x8*)&Bst[ns * 16 + l16][hi * 8];
            b.h[1] = *(const bf16x8*)&Bst[ns * 16 + l16][hi * 8 + 16];
            acc[ns] = __builtin_amdgcn_wmma_f32_16x16x32_bf16(
                false, a.v, false, b.v, (short)0, acc[ns], false, false);
        }
        __syncthreads();
    }

#pragma unroll
    for (int ns = 0; ns < 4; ++ns) {
        const float bvv = bo[n0 + ns * 16 + l16];
#pragma unroll
        for (int r = 0; r < 8; ++r) {
            const int m = m0 + wid * 16 + r + hi * 8;
            out[(size_t)m * H + n0 + ns * 16 + l16] = acc[ns][r] + bvv;
        }
    }
}

// ---------------------------------------------------------------------------
// Launch
// ---------------------------------------------------------------------------
extern "C" void kernel_launch(void* const* d_in, const int* in_sizes, int n_in,
                              void* d_out, int out_size, void* d_ws, size_t ws_size,
                              hipStream_t stream) {
    const float* x   = (const float*)d_in[0];
    const float* Wq  = (const float*)d_in[1];
    const float* bq  = (const float*)d_in[2];
    const float* Wk  = (const float*)d_in[3];
    const float* bk  = (const float*)d_in[4];
    const float* Wv  = (const float*)d_in[5];
    const float* bv  = (const float*)d_in[6];
    const float* Wo  = (const float*)d_in[7];
    const float* bo  = (const float*)d_in[8];
    const int*   pos = (const int*)d_in[9];
    float* out = (float*)d_out;

    unsigned short* ws = (unsigned short*)d_ws;
    size_t o = 0;
    unsigned short* xb    = ws + o; o += (size_t)MTOK * H;   // x bf16
    unsigned short* Wqb   = ws + o; o += (size_t)H * H;
    unsigned short* Wkb   = ws + o; o += (size_t)H * H;
    unsigned short* Wvb   = ws + o; o += (size_t)H * H;
    unsigned short* Wob   = ws + o; o += (size_t)H * H;
    unsigned short* qhb   = ws + o; o += (size_t)MTOK * H;   // head-major q
    unsigned short* khb   = ws + o; o += (size_t)MTOK * H;   // head-major k
    unsigned short* vhtb  = ws + o; o += (size_t)MTOK * H;   // d-major (transposed) v
    unsigned short* attnb = ws + o; o += (size_t)MTOK * H;   // attn output bf16

    const int nx = MTOK * H, nw = H * H;
    mha_cvt_bf16_kernel<<<(nx + 255) / 256, 256, 0, stream>>>(x, xb, nx);
    mha_cvt_bf16_kernel<<<(nw + 255) / 256, 256, 0, stream>>>(Wq, Wqb, nw);
    mha_cvt_bf16_kernel<<<(nw + 255) / 256, 256, 0, stream>>>(Wk, Wkb, nw);
    mha_cvt_bf16_kernel<<<(nw + 255) / 256, 256, 0, stream>>>(Wv, Wvb, nw);
    mha_cvt_bf16_kernel<<<(nw + 255) / 256, 256, 0, stream>>>(Wo, Wob, nw);

    mha_qkv_gemm_rope_kernel<<<dim3(H / 64, MTOK / 128, 3), 256, 0, stream>>>(
        xb, Wqb, Wkb, Wvb, bq, bk, bv, pos, qhb, khb, vhtb);

    mha_attn_kernel<<<dim3(S / 16, NH, BATCH), 128, 0, stream>>>(qhb, khb, vhtb, attnb);

    mha_out_gemm_kernel<<<dim3(H / 64, MTOK / 128), 256, 0, stream>>>(attnb, Wob, bo, out);
}
